// Token_performer_652835029388
// MI455X (gfx1250) — matrix operand order repaired and checked by reading, not compiled, not run on hardware
//
#include <hip/hip_runtime.h>
#include <hip/hip_bf16.h>
#include <math.h>

// ---------------------------------------------------------------------------
// Token-Performer block on MI455X (gfx1250, wave32).
// GEMMs: v_wmma_f32_16x16x32_bf16, 128x128 block tile, triple-buffered LDS
// fed by global_load_async_to_lds_b128, pipeline unrolled by 3 so all buffer
// indices are compile-time constants in steady state.
// ---------------------------------------------------------------------------

typedef __attribute__((ext_vector_type(16))) __bf16        v16bf;
typedef __attribute__((ext_vector_type(8)))  float         v8f;
typedef __attribute__((ext_vector_type(4)))  unsigned int  u32x4;

union ABfrag { v16bf v; u32x4 q[2]; };

// problem constants
#define BB   16
#define TT   3136
#define DIMC 512
#define EMBC 512
#define MRF  256
#define BT   (BB * TT)   // 50176

// GEMM tiling
#define SASTR 40                 // LDS row stride, elements (80B = 16B-aligned)
#define BUFEL (128 * SASTR)      // elements per LDS buffer (10,240 B)

// epilogue modes
#define EP_KQV   0
#define EP_PRM_K 1
#define EP_PRM_Q 2
#define EP_KPTV  3
#define EP_Y     4
#define EP_PROJ  5
#define EP_GELU  6
#define EP_MLP2  7

// CDNA5 async global->LDS copy (16B per lane), ASYNCcnt tracked.
__device__ __forceinline__ void async_copy_b128(unsigned lds_off,
                                                unsigned long long gaddr)
{
    asm volatile("global_load_async_to_lds_b128 %0, %1, off"
                 :: "v"(lds_off), "v"(gaddr) : "memory");
}
#define ASYNC_WAIT_LE4() asm volatile("s_wait_asynccnt 0x4" ::: "memory")
#define ASYNC_WAIT_0()   asm volatile("s_wait_asynccnt 0x0" ::: "memory")

// ---------------------------------------------------------------------------
// Templated bf16 WMMA GEMM:  C[row, col] = sum_k A[row,k] * Bt[col,k]
// ---------------------------------------------------------------------------
template <int EP>
__global__ __launch_bounds__(256) void gemm_bf16_wmma(
    const __bf16* __restrict__ A, const __bf16* __restrict__ Bt,
    const float* __restrict__ bias, const float* __restrict__ extra,
    const float* __restrict__ resid,
    __bf16* __restrict__ obf0, __bf16* __restrict__ obf1,
    __bf16* __restrict__ obf2, float* __restrict__ of32,
    int Mrows, int N, int K, int Tloc,
    long bsA, long bsB, long bsO, long bsE)
{
    const int bz = blockIdx.z;
    A  += (size_t)bz * (size_t)bsA;
    Bt += (size_t)bz * (size_t)bsB;

    __shared__ __bf16 sA[3 * BUFEL];   // 3 x 128x32 bf16
    __shared__ __bf16 sB[3 * BUFEL];

    const int tid  = threadIdx.x;
    const int lane = tid & 31;
    const int wave = tid >> 5;
    const int wm = wave >> 1;            // 0..3  (32-row strip)
    const int wn = wave & 1;             // 0..1  (64-col strip)
    const int rowBase = blockIdx.y * 128;
    const int colBase = blockIdx.x * 128;

    // ---- per-thread async-copy assignments (16B chunks) -------------------
    // Each 128x32 tile = 512 chunks -> 2 per thread per matrix.
    const int c1 = tid + 256;
    const int r0 = tid >> 2, k0 = (tid & 3) << 3;
    const int r1 = c1  >> 2, k1 = (c1  & 3) << 3;
    int ga0r = rowBase + r0; if (ga0r >= Mrows) ga0r = Mrows - 1;  // clamp:
    int ga1r = rowBase + r1; if (ga1r >= Mrows) ga1r = Mrows - 1;  // OOB rows feed skipped C rows only

    // running per-lane global addresses, advanced 64B per K-tile
    unsigned long long vA0 = (unsigned long long)(A  + (size_t)ga0r * K + k0);
    unsigned long long vA1 = (unsigned long long)(A  + (size_t)ga1r * K + k1);
    unsigned long long vB0 = (unsigned long long)(Bt + (size_t)(colBase + r0) * K + k0);
    unsigned long long vB1 = (unsigned long long)(Bt + (size_t)(colBase + r1) * K + k1);

    // LDS byte offsets per buffer (flat-pointer low 32 bits == LDS offset)
    unsigned dA0[3], dA1[3], dB0[3], dB1[3];
#pragma unroll
    for (int b = 0; b < 3; ++b) {
        dA0[b] = (unsigned)(unsigned long long)(&sA[b * BUFEL + r0 * SASTR + k0]);
        dA1[b] = (unsigned)(unsigned long long)(&sA[b * BUFEL + r1 * SASTR + k1]);
        dB0[b] = (unsigned)(unsigned long long)(&sB[b * BUFEL + r0 * SASTR + k0]);
        dB1[b] = (unsigned)(unsigned long long)(&sB[b * BUFEL + r1 * SASTR + k1]);
    }

    auto issue = [&](int tb) {
        async_copy_b128(dA0[tb], vA0);
        async_copy_b128(dA1[tb], vA1);
        async_copy_b128(dB0[tb], vB0);
        async_copy_b128(dB1[tb], vB1);
        vA0 += 64; vA1 += 64; vB0 += 64; vB1 += 64;
    };

    v8f acc[2][4];
#pragma unroll
    for (int i = 0; i < 2; ++i)
#pragma unroll
        for (int j = 0; j < 4; ++j)
#pragma unroll
            for (int r = 0; r < 8; ++r) acc[i][j][r] = 0.0f;

    // fragment LDS offsets (CDNA5 16-bit WMMA lane layouts)
    const int kAo = (lane & 16) ? 8 : 0;     // A: hi lanes hold K 8..15 / 24..31
    const int kBo = (lane & 16) ? 8 : 0;     // B: hi lanes hold K 8..15 / 24..31
    const int aOff = (wm * 32 + (lane & 15)) * SASTR + kAo;
    const int bOff = (wn * 64 + (lane & 15)) * SASTR + kBo;

    auto compute = [&](int off) {            // off = buf * BUFEL (elements)
        ABfrag bf[4];
#pragma unroll
        for (int ni = 0; ni < 4; ++ni) {
            const __bf16* pb = sB + off + bOff + ni * 16 * SASTR;
            bf[ni].q[0] = *reinterpret_cast<const u32x4*>(pb);
            bf[ni].q[1] = *reinterpret_cast<const u32x4*>(pb + 16);
        }
#pragma unroll
        for (int mi = 0; mi < 2; ++mi) {
            ABfrag af;
            const __bf16* pa = sA + off + aOff + mi * 16 * SASTR;
            af.q[0] = *reinterpret_cast<const u32x4*>(pa);
            af.q[1] = *reinterpret_cast<const u32x4*>(pa + 16);
#pragma unroll
            for (int ni = 0; ni < 4; ++ni)
                acc[mi][ni] = __builtin_amdgcn_wmma_f32_16x16x32_bf16(
                    false, af.v, false, bf[ni].v, (short)0, acc[mi][ni],
                    false, false);
        }
    };

    // ---- software pipeline: depth 2, 3 buffers, unrolled x3 ---------------
    const int ntiles  = K >> 5;              // always >= 8 here
    const int nfull   = ntiles - 2;          // iterations that also issue
    const int nsteady = (nfull / 3) * 3;

    issue(0);
    issue(1);

    int i = 0;
    for (; i < nsteady; i += 3) {
        ASYNC_WAIT_LE4(); __syncthreads(); issue(2); compute(0 * BUFEL);
        ASYNC_WAIT_LE4(); __syncthreads(); issue(0); compute(1 * BUFEL);
        ASYNC_WAIT_LE4(); __syncthreads(); issue(1); compute(2 * BUFEL);
    }
    for (; i < ntiles; ++i) {                // tiny tail (<= 4 iterations)
        if (i < nfull)           { ASYNC_WAIT_LE4(); __syncthreads(); issue((i + 2) % 3); }
        else if (i < ntiles - 1) { ASYNC_WAIT_LE4(); __syncthreads(); }
        else                     { ASYNC_WAIT_0();   __syncthreads(); }
        compute((i % 3) * BUFEL);
    }

    // ---- epilogue: C f32 16x16 layout: lanes 0-15 N=lane, M=r;
    //      lanes 16-31 N=lane-16, M=8+r ------------------------------------
#pragma unroll
    for (int mi = 0; mi < 2; ++mi)
#pragma unroll
        for (int ni = 0; ni < 4; ++ni)
#pragma unroll
            for (int r = 0; r < 8; ++r) {
                int row = rowBase + wm * 32 + mi * 16 + ((lane & 16) ? 8 + r : r);
                int col = colBase + wn * 64 + ni * 16 + (lane & 15);
                if (row >= Mrows || col >= N) continue;
                float v = acc[mi][ni][r];

                if (EP == EP_KQV) {
                    v += bias[col];
                    int b = row / Tloc;
                    int t = row - b * Tloc;
                    if (col < EMBC) {
                        obf0[(size_t)row * EMBC + col] = (__bf16)v;
                    } else if (col < 2 * EMBC) {
                        obf1[(size_t)row * EMBC + (col - EMBC)] = (__bf16)v;
                    } else {
                        obf2[((size_t)b * EMBC + (col - 2 * EMBC)) * Tloc + t] = (__bf16)v;
                    }
                } else if (EP == EP_PRM_K) {
                    v = __expf(v - extra[row]) * 0.0625f;   // 1/sqrt(256)
                    int b = row / Tloc;
                    int t = row - b * Tloc;
                    obf0[((size_t)b * MRF + col) * Tloc + t] = (__bf16)v;   // kpT
                } else if (EP == EP_PRM_Q) {
                    v = __expf(v - extra[row]) * 0.0625f;
                    obf0[(size_t)row * MRF + col] = (__bf16)v;              // qp
                } else if (EP == EP_KPTV) {
                    obf0[(size_t)bz * bsO + (size_t)row * N + col] = (__bf16)v;
                } else if (EP == EP_Y) {
                    float d = extra[(size_t)bz * bsE + row];
                    v = v / (d + 1e-8f);
                    obf0[(size_t)bz * bsO + (size_t)row * N + col] = (__bf16)v;
                } else if (EP == EP_PROJ) {
                    v += bias[col] + resid[(size_t)row * N + col];
                    of32[(size_t)row * N + col] = v;
                } else if (EP == EP_GELU) {
                    v += bias[col];
                    v = 0.5f * v * (1.0f + erff(v * 0.70710678118654752f));
                    obf0[(size_t)row * N + col] = (__bf16)v;
                } else { // EP_MLP2
                    v += bias[col] + resid[(size_t)row * N + col];
                    of32[(size_t)row * N + col] = v;
                }
            }
}

// ---------------------------------------------------------------------------
// Support kernels
// ---------------------------------------------------------------------------
__global__ __launch_bounds__(256) void ln_to_bf16_kernel(
    const float* __restrict__ x, const float* __restrict__ g,
    const float* __restrict__ b, __bf16* __restrict__ out, int rows)
{
    int lane = threadIdx.x & 31;
    int row  = blockIdx.x * 8 + (threadIdx.x >> 5);
    if (row >= rows) return;
    const float* xr = x + (size_t)row * DIMC;
    float vals[16];
    float s = 0.f, ss = 0.f;
#pragma unroll
    for (int j = 0; j < 16; ++j) {
        float v = xr[lane + j * 32];
        vals[j] = v; s += v; ss += v * v;
    }
#pragma unroll
    for (int m = 16; m >= 1; m >>= 1) {
        s  += __shfl_xor(s,  m, 32);
        ss += __shfl_xor(ss, m, 32);
    }
    float mu  = s * (1.0f / DIMC);
    float var = ss * (1.0f / DIMC) - mu * mu;
    float rs  = rsqrtf(var + 1e-5f);
    __bf16* orow = out + (size_t)row * DIMC;
#pragma unroll
    for (int j = 0; j < 16; ++j) {
        int c = lane + j * 32;
        orow[c] = (__bf16)((vals[j] - mu) * rs * g[c] + b[c]);
    }
}

// xd[row] = 0.5 * sum_i in[row,i]^2    (512 cols, bf16 input)
__global__ __launch_bounds__(256) void rowhalfsq_kernel(
    const __bf16* __restrict__ in, float* __restrict__ out, int rows)
{
    int lane = threadIdx.x & 31;
    int row  = blockIdx.x * 8 + (threadIdx.x >> 5);
    if (row >= rows) return;
    const __bf16* r = in + (size_t)row * EMBC;
    float s = 0.f;
#pragma unroll
    for (int j = 0; j < 16; ++j) {
        float v = (float)r[lane + j * 32];
        s += v * v;
    }
#pragma unroll
    for (int m = 16; m >= 1; m >>= 1) s += __shfl_xor(s, m, 32);
    if (lane == 0) out[row] = 0.5f * s;
}

// kpsum[b*M+m] = sum_t kpT[b][m][t]
__global__ __launch_bounds__(256) void kpsum_kernel(
    const __bf16* __restrict__ kpT, float* __restrict__ out, int rows, int len)
{
    int lane = threadIdx.x & 31;
    int row  = blockIdx.x * 8 + (threadIdx.x >> 5);
    if (row >= rows) return;
    const __bf16* r = kpT + (size_t)row * len;
    float s = 0.f;
    for (int j = lane; j < len; j += 32) s += (float)r[j];
#pragma unroll
    for (int m = 16; m >= 1; m >>= 1) s += __shfl_xor(s, m, 32);
    if (lane == 0) out[row] = s;
}

// D[row] = sum_m qp[row,m] * kpsum[b,m]
__global__ __launch_bounds__(256) void dcalc_kernel(
    const __bf16* __restrict__ qp, const float* __restrict__ kpsum,
    float* __restrict__ D, int rows)
{
    int lane = threadIdx.x & 31;
    int row  = blockIdx.x * 8 + (threadIdx.x >> 5);
    if (row >= rows) return;
    int b = row / TT;
    const __bf16* qr = qp + (size_t)row * MRF;
    const float*  ks = kpsum + (size_t)b * MRF;
    float s = 0.f;
#pragma unroll
    for (int j = 0; j < 8; ++j) {
        int m = lane + j * 32;
        s += (float)qr[m] * ks[m];
    }
#pragma unroll
    for (int m = 16; m >= 1; m >>= 1) s += __shfl_xor(s, m, 32);
    if (lane == 0) D[row] = s;
}

// dst[n*K + k] = (bf16) src[k*N + n]   (weight transpose-convert)
__global__ __launch_bounds__(256) void cvt_transpose_kernel(
    const float* __restrict__ src, __bf16* __restrict__ dst, int K, int N)
{
    size_t i = (size_t)blockIdx.x * 256 + threadIdx.x;
    if (i >= (size_t)K * N) return;
    size_t n = i / (size_t)K;
    size_t k = i - n * (size_t)K;
    dst[i] = (__bf16)src[k * (size_t)N + n];
}

__global__ __launch_bounds__(256) void cvt_plain_kernel(
    const float* __restrict__ src, __bf16* __restrict__ dst, size_t n)
{
    size_t i = (size_t)blockIdx.x * 256 + threadIdx.x;
    if (i < n) dst[i] = (__bf16)src[i];
}

// ---------------------------------------------------------------------------
// Host-side orchestration
// ---------------------------------------------------------------------------
extern "C" void kernel_launch(void* const* d_in, const int* in_sizes, int n_in,
                              void* d_out, int out_size, void* d_ws, size_t ws_size,
                              hipStream_t stream)
{
    (void)in_sizes; (void)n_in; (void)out_size; (void)ws_size;

    const float* x      = (const float*)d_in[0];
    const float* w_kqv  = (const float*)d_in[1];
    const float* b_kqv  = (const float*)d_in[2];
    const float* w_proj = (const float*)d_in[3];
    const float* b_proj = (const float*)d_in[4];
    const float* g1     = (const float*)d_in[5];
    const float* be1    = (const float*)d_in[6];
    const float* g2     = (const float*)d_in[7];
    const float* be2    = (const float*)d_in[8];
    const float* w_mlp1 = (const float*)d_in[9];
    const float* b_mlp1 = (const float*)d_in[10];
    const float* w_mlp2 = (const float*)d_in[11];
    const float* b_mlp2 = (const float*)d_in[12];
    const float* w_rand = (const float*)d_in[13];
    float* out = (float*)d_out;

    // --- workspace carving (deterministic) ---
    char* ws = (char*)d_ws;
    size_t off = 0;
    auto take = [&](size_t bytes) -> char* {
        char* p = ws + off;
        off += (bytes + 255) & ~(size_t)255;
        return p;
    };
    __bf16* wkqvT  = (__bf16*)take((size_t)1536 * 512 * 2);
    __bf16* wrandB = (__bf16*)take((size_t)MRF * EMBC * 2);
    __bf16* wprojT = (__bf16*)take((size_t)512 * 512 * 2);
    __bf16* wmlp1T = (__bf16*)take((size_t)512 * 512 * 2);
    __bf16* wmlp2T = (__bf16*)take((size_t)512 * 512 * 2);
    __bf16* h1     = (__bf16*)take((size_t)BT * 512 * 2);   // LN1 out; reused for y
    __bf16* k_bf   = (__bf16*)take((size_t)BT * 512 * 2);   // reused for h2
    __bf16* q_bf   = (__bf16*)take((size_t)BT * 512 * 2);   // reused for h3
    __bf16* vT     = (__bf16*)take((size_t)BT * 512 * 2);
    float*  xd_k   = (float*)take((size_t)BT * 4);
    float*  xd_q   = (float*)take((size_t)BT * 4);
    __bf16* kpT    = (__bf16*)take((size_t)BB * MRF * TT * 2);
    __bf16* qp     = (__bf16*)take((size_t)BT * MRF * 2);
    float*  kpsum  = (float*)take((size_t)BB * MRF * 4);
    float*  Dbuf   = (float*)take((size_t)BT * 4);
    __bf16* kptv   = (__bf16*)take((size_t)BB * EMBC * MRF * 2);
    float*  x2     = (float*)take((size_t)BT * 512 * 4);
    __bf16* y_bf = h1;     // h1 dead after kqv GEMM
    __bf16* h2   = k_bf;   // k dead after prm_k GEMM
    __bf16* h3   = q_bf;   // q dead after prm_q GEMM

    const dim3 blk(256);

    // 1) weight conversion
    cvt_transpose_kernel<<<(1536 * 512 + 255) / 256, blk, 0, stream>>>(w_kqv,  wkqvT,  512, 1536);
    cvt_transpose_kernel<<<(512 * 512 + 255) / 256, blk, 0, stream>>>(w_proj, wprojT, 512, 512);
    cvt_transpose_kernel<<<(512 * 512 + 255) / 256, blk, 0, stream>>>(w_mlp1, wmlp1T, 512, 512);
    cvt_transpose_kernel<<<(512 * 512 + 255) / 256, blk, 0, stream>>>(w_mlp2, wmlp2T, 512, 512);
    cvt_plain_kernel<<<(MRF * EMBC + 255) / 256, blk, 0, stream>>>(w_rand, wrandB, (size_t)MRF * EMBC);

    // 2) LN1
    ln_to_bf16_kernel<<<(BT + 7) / 8, blk, 0, stream>>>(x, g1, be1, h1, BT);

    // 3) kqv GEMM: [BT,512] x Bt[1536,512] -> split k / q / vT
    gemm_bf16_wmma<EP_KQV><<<dim3(1536 / 128, BT / 128, 1), blk, 0, stream>>>(
        h1, wkqvT, b_kqv, nullptr, nullptr, k_bf, q_bf, vT, nullptr,
        BT, 1536, 512, TT, 0, 0, 0, 0);

    // 4) xd = 0.5*||row||^2
    rowhalfsq_kernel<<<(BT + 7) / 8, blk, 0, stream>>>(k_bf, xd_k, BT);
    rowhalfsq_kernel<<<(BT + 7) / 8, blk, 0, stream>>>(q_bf, xd_q, BT);

    // 5) kp = exp(k@Wr^T - xd)/sqrt(M), stored transposed kpT[b][m][t]
    gemm_bf16_wmma<EP_PRM_K><<<dim3(MRF / 128, BT / 128, 1), blk, 0, stream>>>(
        k_bf, wrandB, nullptr, xd_k, nullptr, kpT, nullptr, nullptr, nullptr,
        BT, MRF, 512, TT, 0, 0, 0, 0);

    // 6) qp (row-major)
    gemm_bf16_wmma<EP_PRM_Q><<<dim3(MRF / 128, BT / 128, 1), blk, 0, stream>>>(
        q_bf, wrandB, nullptr, xd_q, nullptr, qp, nullptr, nullptr, nullptr,
        BT, MRF, 512, TT, 0, 0, 0, 0);

    // 7) kpsum[b][m] = sum_t kpT
    kpsum_kernel<<<(BB * MRF + 7) / 8, blk, 0, stream>>>(kpT, kpsum, BB * MRF, TT);

    // 8) D[b*T+t] = qp . kpsum
    dcalc_kernel<<<(BT + 7) / 8, blk, 0, stream>>>(qp, kpsum, Dbuf, BT);

    // 9) kptv[b][n][m] = sum_t vT[b][n][t] * kpT[b][m][t]   (batched, K=3136)
    gemm_bf16_wmma<EP_KPTV><<<dim3(MRF / 128, EMBC / 128, BB), blk, 0, stream>>>(
        vT, kpT, nullptr, nullptr, nullptr, kptv, nullptr, nullptr, nullptr,
        EMBC, MRF, TT, TT,
        (long)EMBC * TT, (long)MRF * TT, (long)EMBC * MRF, 0);

    // 10) y[b][t][n] = (qp @ kptv^T) / (D+eps)  (batched, K=256)
    gemm_bf16_wmma<EP_Y><<<dim3(EMBC / 128, (TT + 127) / 128, BB), blk, 0, stream>>>(
        qp, kptv, nullptr, Dbuf, nullptr, y_bf, nullptr, nullptr, nullptr,
        TT, EMBC, MRF, TT,
        (long)TT * MRF, (long)EMBC * MRF, (long)TT * EMBC, (long)TT);

    // 11) x2 = x + y @ w_proj + b_proj   (fp32 residual stream)
    gemm_bf16_wmma<EP_PROJ><<<dim3(EMBC / 128, BT / 128, 1), blk, 0, stream>>>(
        y_bf, wprojT, b_proj, nullptr, x, nullptr, nullptr, nullptr, x2,
        BT, EMBC, EMBC, TT, 0, 0, 0, 0);

    // 12) LN2
    ln_to_bf16_kernel<<<(BT + 7) / 8, blk, 0, stream>>>(x2, g2, be2, h2, BT);

    // 13) h3 = gelu(h2 @ w_mlp1 + b_mlp1)
    gemm_bf16_wmma<EP_GELU><<<dim3(EMBC / 128, BT / 128, 1), blk, 0, stream>>>(
        h2, wmlp1T, b_mlp1, nullptr, nullptr, h3, nullptr, nullptr, nullptr,
        BT, EMBC, EMBC, TT, 0, 0, 0, 0);

    // 14) out = x2 + h3 @ w_mlp2 + b_mlp2
    gemm_bf16_wmma<EP_MLP2><<<dim3(EMBC / 128, BT / 128, 1), blk, 0, stream>>>(
        h3, wmlp2T, b_mlp2, nullptr, x2, nullptr, nullptr, nullptr, out,
        BT, EMBC, EMBC, TT, 0, 0, 0, 0);
}
